// DAMIC_84705345012347
// MI455X (gfx1250) — matrix operand order repaired and checked by reading, not compiled, not run on hardware
//
#include <hip/hip_runtime.h>
#include <hip/hip_bf16.h>
#include <math.h>

// ---------- types ----------
typedef __bf16 v16bf __attribute__((ext_vector_type(16)));
typedef __bf16 v8bf  __attribute__((ext_vector_type(8)));
typedef float  v8f   __attribute__((ext_vector_type(8)));

union ABf { v16bf v; v8bf h[2]; };

__device__ __forceinline__ v8f wmma_bf16(const ABf& a, const ABf& b, v8f c) {
  return __builtin_amdgcn_wmma_f32_16x16x32_bf16(false, a.v, false, b.v, (short)0, c, false, false);
}
__device__ __forceinline__ float sigf(float x) { return 1.f / (1.f + __expf(-x)); }

// CDNA5 async global->LDS copy (16B per lane), tracked by ASYNCcnt.
__device__ __forceinline__ void async_g2l_b128(unsigned lds_byte_off, const void* gaddr) {
  asm volatile("global_load_async_to_lds_b128 %0, %1, off"
               :: "v"(lds_byte_off), "v"(gaddr) : "memory");
}
__device__ __forceinline__ void async_wait0() {
  asm volatile("s_wait_asynccnt 0x0" ::: "memory");
}

// fragment loaders (address space inferred at inline site: LDS or global)
__device__ __forceinline__ void loadA(const __bf16* aprow, int kc, ABf& A) {
  const __bf16* ap = aprow + kc * 32;
  A.h[0] = *(const v8bf*)(ap);
  A.h[1] = *(const v8bf*)(ap + 16);
}
__device__ __forceinline__ void loadB(const __bf16* bbase, int lane, int kc, ABf& B) {
  const __bf16* bp = bbase + (size_t)(kc * 32 + lane) * 16;
  B.h[0] = *(const v8bf*)(bp);
  B.h[1] = *(const v8bf*)(bp + 8);
}

// two-stage software-pipelined K loop, ping-pong regs, no in-loop branches.
// kc_n >= 2; odd kc_n overreads exactly one chunk (buffers padded for this).
#define K_PIPELINE(acc, aprow, bbase, lane, kc_n)                         \
  {                                                                       \
    ABf a0_, b0_, a1_, b1_;                                               \
    loadA(aprow, 0, a0_); loadB(bbase, lane, 0, b0_);                     \
    loadA(aprow, 1, a1_); loadB(bbase, lane, 1, b1_);                     \
    int kc_ = 0;                                                          \
    _Pragma("unroll 1")                                                   \
    for (; kc_ + 2 < (kc_n); kc_ += 2) {                                  \
      acc = wmma_bf16(a0_, b0_, acc);                                     \
      loadA(aprow, kc_ + 2, a0_); loadB(bbase, lane, kc_ + 2, b0_);       \
      acc = wmma_bf16(a1_, b1_, acc);                                     \
      loadA(aprow, kc_ + 3, a1_); loadB(bbase, lane, kc_ + 3, b1_);       \
    }                                                                     \
    acc = wmma_bf16(a0_, b0_, acc);                                       \
    if (kc_ + 1 < (kc_n)) acc = wmma_bf16(a1_, b1_, acc);                 \
  }

// ---------- model dims ----------
#define BATCH 32
#define TSEQ  64
#define LSEQ  48
#define EMBD  300
#define EP    304      // padded embedding stride (mult of 8)
#define LP    54       // padded rows in LDS x-tile (covers K overreads)
#define NFILT 100
#define HID   256
#define DPAD  320      // padded LSTM input dim (300 -> 320)
#define GATES 1024     // 4*HID
#define BT    2048     // BATCH*TSEQ
#define OUTD  32

// =====================================================================
// Pack f32 weights into exact per-lane WMMA bf16 B-fragments.
// layout: frag[((nt*KC + kc)*32 + lane)*16 + i], B[k][n], k=kc*32+(lane>>4)*8+(i&7)+(i>>3)*16
// mode 0: src is [Nsrc rows][Ksrc cols], B[k][n] = src[n*Ksrc + k]
// mode 1: conv (NF,EMB,fs): k -> (r=k/304, e=k%304), B[k][n]=src[(n*300+e)*fs+r]
// =====================================================================
__global__ void __launch_bounds__(32) damic_packB(
    const float* __restrict__ src, __bf16* __restrict__ dst,
    int KC, int mode, int Nsrc, int Ksrc, int fs)
{
  int nt = blockIdx.x, kc = blockIdx.y;
  int lane = threadIdx.x;
  int n = nt * 16 + (lane & 15);
  int kbase = kc * 32 + (lane >> 4) * 8;
  __bf16* d = dst + ((size_t)(nt * KC + kc) * 32 + lane) * 16;
  #pragma unroll 1
  for (int i = 0; i < 16; ++i) {
    int k = kbase + (i & 7) + (i >> 3) * 16;
    float v = 0.f;
    if (mode == 0) {
      if (n < Nsrc && k < Ksrc) v = src[(size_t)n * Ksrc + k];
    } else {
      int e = k % EP, r = k / EP;
      if (n < NFILT && e < EMBD && r < fs) v = src[((size_t)n * EMBD + e) * fs + r];
    }
    d[i] = (__bf16)v;
  }
}

// =====================================================================
// Fused embedding-gather (async global->LDS) + 3x conv-as-GEMM + ReLU +
// max-pool. One block per (b,t) sequence; Toeplitz A tile in LDS (bf16).
// Output: XT[t][b][320] bf16 (cols 0..99 fs3 | 100..199 fs4 | 200..299 fs5)
// =====================================================================
__global__ void __launch_bounds__(256) damic_conv(
    const int* __restrict__ dialogue, const float* __restrict__ emb,
    const __bf16* __restrict__ c3f, const __bf16* __restrict__ c4f,
    const __bf16* __restrict__ c5f,
    const float* __restrict__ cb3, const float* __restrict__ cb4,
    const float* __restrict__ cb5,
    __bf16* __restrict__ XT)
{
  __shared__ float  stage[LSEQ * EMBD];   // 57.6 KB raw gathered f32 rows
  __shared__ __bf16 xs[LP * EP];          // 32.8 KB padded bf16 Toeplitz tile
  const int bt  = blockIdx.x;
  const int tid = threadIdx.x;

  // --- phase 1: async-DMA the 48 gathered embedding rows into LDS ---
  // 48 rows * 1200 B = 3600 16-byte chunks (rows are 16B aligned).
  const unsigned stage_base = (unsigned)(unsigned long long)&stage[0];
  for (int c = tid; c < LSEQ * 75; c += 256) {
    int l = c / 75, chunk = c % 75;
    int tok = dialogue[bt * LSEQ + l];
    const char* g = (const char*)(emb + (size_t)tok * EMBD) + chunk * 16;
    async_g2l_b128(stage_base + (unsigned)(l * 1200 + chunk * 16), g);
  }
  async_wait0();
  __syncthreads();

  // --- phase 2: f32 -> bf16 into padded tile (pad rows/cols = 0) ---
  for (int i = tid; i < LP * EP; i += 256) {
    int l = i / EP, e = i % EP;
    float v = (l < LSEQ && e < EMBD) ? stage[l * EMBD + e] : 0.f;
    xs[i] = (__bf16)v;
  }
  __syncthreads();

  const int t = bt & (TSEQ - 1), b = bt >> 6;
  __bf16* xtrow = XT + ((size_t)t * BATCH + b) * DPAD;
  if (tid < DPAD - 3 * NFILT) xtrow[3 * NFILT + tid] = (__bf16)0.f;

  const int lane   = tid & 31;
  const int wave   = tid >> 5;
  const int col    = lane & 15;
  const int rowoff = (lane >> 4) * 8;
  const int k0off  = (lane >> 4) * 8;

  for (int job = __builtin_amdgcn_readfirstlane(wave); job < 21; job += 8) {
    const int fsi = job / 7, nt = job % 7;     // scalar (job is scalar)
    int kc_n, outlen;
    const __bf16* bbase;
    const float* cbp;
    if (fsi == 0)      { kc_n = 29; outlen = 46; bbase = c3f + (size_t)nt * 29 * 512; cbp = cb3; }
    else if (fsi == 1) { kc_n = 38; outlen = 45; bbase = c4f + (size_t)nt * 38 * 512; cbp = cb4; }
    else               { kc_n = 48; outlen = 44; bbase = c5f + (size_t)nt * 48 * 512; cbp = cb5; }
    const int f = nt * 16 + col;
    const float cb = (f < NFILT) ? cbp[f] : 0.f;
    float mx = 0.f;                               // ReLU => max >= 0
    #pragma unroll 1
    for (int mt = 0; mt < 3; ++mt) {
      const __bf16* aprow = xs + (mt * 16 + col) * EP + k0off;
      v8f acc = {};
      K_PIPELINE(acc, aprow, bbase, lane, kc_n);
      #pragma unroll
      for (int v = 0; v < 8; ++v) {
        int m = mt * 16 + rowoff + v;
        if (m < outlen) mx = fmaxf(mx, fmaxf(acc[v] + cb, 0.f));
      }
    }
    mx = fmaxf(mx, __shfl_xor(mx, 16, 32));       // combine M=8 half-lanes
    if (lane < 16 && f < NFILT) xtrow[fsi * NFILT + f] = (__bf16)mx;
  }
}

// =====================================================================
// LSTM input projection: XG[dir][t*32+b][1024] = XT @ w_ih^T + (b_ih+b_hh)
// one wave per (mt, nt, dir) tile; K = 320 (10 chunks), pipelined
// =====================================================================
__global__ void __launch_bounds__(32) damic_xg(
    const __bf16* __restrict__ XT,
    const __bf16* __restrict__ wihf, const __bf16* __restrict__ wihr,
    const float* __restrict__ bihf, const float* __restrict__ bhhf,
    const float* __restrict__ bihr, const float* __restrict__ bhhr,
    float* __restrict__ XG)
{
  const int mt = blockIdx.x, nt = blockIdx.y, dir = blockIdx.z;
  const int lane = threadIdx.x;
  const int col = lane & 15, rowoff = (lane >> 4) * 8, k0off = (lane >> 4) * 8;
  const __bf16* wf;
  if (dir) wf = wihr + (size_t)nt * 10 * 512; else wf = wihf + (size_t)nt * 10 * 512;
  const __bf16* arow = XT + (size_t)(mt * 16 + col) * DPAD + k0off;
  v8f acc = {};
  K_PIPELINE(acc, arow, wf, lane, 10);
  const int n = nt * 16 + col;
  const float bias = dir ? (bihr[n] + bhhr[n]) : (bihf[n] + bhhf[n]);
  float* out = XG + (size_t)dir * BT * GATES;
  #pragma unroll
  for (int v = 0; v < 8; ++v)
    out[(size_t)(mt * 16 + rowoff + v) * GATES + n] = acc[v] + bias;
}

// =====================================================================
// Recurrent scan: one 1024-thread block per direction (32 waves cover
// the 2x64 WMMA tile grid of the 32x1024 gate GEMM each step).
// gates in 128KB LDS (f32), h in 16KB LDS (bf16), c in registers.
// HCAT[b][t][dir*256 + j] bf16
// =====================================================================
__global__ void __launch_bounds__(1024) damic_lstm(
    const float* __restrict__ XG,
    const __bf16* __restrict__ whhf, const __bf16* __restrict__ whhr,
    __bf16* __restrict__ HCAT)
{
  __shared__ float  gates[BATCH * GATES];     // 128 KB
  __shared__ __bf16 hbuf[BATCH * HID + 64];   // 16 KB (+pipeline pad)
  const int dir  = blockIdx.x;
  const int tid  = threadIdx.x;
  const int lane = tid & 31, wave = tid >> 5;
  const int col = lane & 15, rowoff = (lane >> 4) * 8, k0off = (lane >> 4) * 8;
  const __bf16* whh;
  if (dir) whh = whhr; else whh = whhf;
  const float* xg = XG + (size_t)dir * BT * GATES;
  const int wave_s = __builtin_amdgcn_readfirstlane(wave);

  for (int i = tid; i < BATCH * HID + 64; i += 1024) hbuf[i] = (__bf16)0.f;
  float c_reg[8];
  #pragma unroll
  for (int i = 0; i < 8; ++i) c_reg[i] = 0.f;
  __syncthreads();

  #pragma unroll 1
  for (int step = 0; step < TSEQ; ++step) {
    const int tt = dir ? (TSEQ - 1 - step) : step;
    const float* xr = xg + (size_t)tt * BATCH * GATES;
    // gates = h_prev @ w_hh^T + xg[tt]
    #pragma unroll 1
    for (int jj = 0; jj < 4; ++jj) {
      const int job = wave_s + jj * 32;         // scalar: 128 tiles = 2mt*64nt
      const int mt = job >> 6, nt = job & 63;
      const __bf16* bbase = whh + (size_t)nt * 8 * 512;
      const __bf16* aprow = hbuf + (mt * 16 + col) * HID + k0off;
      v8f acc = {};
      K_PIPELINE(acc, aprow, bbase, lane, 8);
      const int n = nt * 16 + col;
      #pragma unroll
      for (int v = 0; v < 8; ++v) {
        int r = mt * 16 + rowoff + v;
        gates[r * GATES + n] = acc[v] + xr[(size_t)r * GATES + n];
      }
    }
    __syncthreads();
    // elementwise cell update (torch gate order i,f,g,o)
    #pragma unroll
    for (int i = 0; i < 8; ++i) {
      const int idx = tid * 8 + i;
      const int b = idx >> 8, j = idx & 255;
      const float gi = gates[b * GATES + j];
      const float gf = gates[b * GATES + HID + j];
      const float gg = gates[b * GATES + 2 * HID + j];
      const float go = gates[b * GATES + 3 * HID + j];
      const float c = sigf(gf) * c_reg[i] + sigf(gi) * tanhf(gg);
      c_reg[i] = c;
      const float h = sigf(go) * tanhf(c);
      hbuf[b * HID + j] = (__bf16)h;
      HCAT[((size_t)b * TSEQ + tt) * (2 * HID) + dir * HID + j] = (__bf16)h;
    }
    __syncthreads();
  }
}

// =====================================================================
// Head: out[b][t][o] = sigmoid(HCAT[b][t][:] @ head_w^T + head_b)
// M=2048 (b*64+t), N=32 (2 tiles), K=512 (16 chunks), pipelined
// =====================================================================
__global__ void __launch_bounds__(32) damic_head(
    const __bf16* __restrict__ HCAT, const __bf16* __restrict__ hwf,
    const float* __restrict__ headb, float* __restrict__ out)
{
  const int mt = blockIdx.x, nt = blockIdx.y;
  const int lane = threadIdx.x;
  const int col = lane & 15, rowoff = (lane >> 4) * 8, k0off = (lane >> 4) * 8;
  const __bf16* bbase = hwf + (size_t)nt * 16 * 512;
  const __bf16* arow = HCAT + (size_t)(mt * 16 + col) * (2 * HID) + k0off;
  v8f acc = {};
  K_PIPELINE(acc, arow, bbase, lane, 16);
  const int n = nt * 16 + col;          // n < 32
  const float bias = headb[n];
  #pragma unroll
  for (int v = 0; v < 8; ++v) {
    float x = acc[v] + bias;
    out[(size_t)(mt * 16 + rowoff + v) * OUTD + n] = sigf(x);
  }
}

// =====================================================================
extern "C" void kernel_launch(void* const* d_in, const int* in_sizes, int n_in,
                              void* d_out, int out_size, void* d_ws, size_t ws_size,
                              hipStream_t stream) {
  (void)in_sizes; (void)n_in; (void)out_size; (void)ws_size;
  const int*   dialogue = (const int*)  d_in[0];
  const float* emb      = (const float*)d_in[1];
  const float* cw3 = (const float*)d_in[2];  const float* cb3 = (const float*)d_in[3];
  const float* cw4 = (const float*)d_in[4];  const float* cb4 = (const float*)d_in[5];
  const float* cw5 = (const float*)d_in[6];  const float* cb5 = (const float*)d_in[7];
  const float* wihf = (const float*)d_in[8];  const float* whhf = (const float*)d_in[9];
  const float* bihf = (const float*)d_in[10]; const float* bhhf = (const float*)d_in[11];
  const float* wihr = (const float*)d_in[12]; const float* whhr = (const float*)d_in[13];
  const float* bihr = (const float*)d_in[14]; const float* bhhr = (const float*)d_in[15];
  const float* headw = (const float*)d_in[16]; const float* headb = (const float*)d_in[17];

  char* ws = (char*)d_ws;
  auto bump = [&](size_t bytes) -> char* {
    char* p = ws; ws += (bytes + 1024 + 255) & ~(size_t)255;  // +1KB pipeline pad
    return p;
  };
  __bf16* XT   = (__bf16*)bump((size_t)BT * DPAD * 2);          // 1.25 MB
  float*  XG   = (float*) bump((size_t)2 * BT * GATES * 4);     // 16 MB
  __bf16* HCAT = (__bf16*)bump((size_t)BT * 2 * HID * 2);       // 2 MB
  __bf16* C3F  = (__bf16*)bump((size_t)7 * 29 * 512 * 2);
  __bf16* C4F  = (__bf16*)bump((size_t)7 * 38 * 512 * 2);
  __bf16* C5F  = (__bf16*)bump((size_t)7 * 48 * 512 * 2);
  __bf16* WIHF = (__bf16*)bump((size_t)64 * 10 * 512 * 2);
  __bf16* WIHR = (__bf16*)bump((size_t)64 * 10 * 512 * 2);
  __bf16* WHHF = (__bf16*)bump((size_t)64 * 8 * 512 * 2);
  __bf16* WHHR = (__bf16*)bump((size_t)64 * 8 * 512 * 2);
  __bf16* HWF  = (__bf16*)bump((size_t)2 * 16 * 512 * 2);

  // pack all GEMM B-operands into per-lane WMMA fragments (bf16)
  damic_packB<<<dim3(7, 29), 32, 0, stream>>>(cw3, C3F, 29, 1, NFILT, 0, 3);
  damic_packB<<<dim3(7, 38), 32, 0, stream>>>(cw4, C4F, 38, 1, NFILT, 0, 4);
  damic_packB<<<dim3(7, 48), 32, 0, stream>>>(cw5, C5F, 48, 1, NFILT, 0, 5);
  damic_packB<<<dim3(64, 10), 32, 0, stream>>>(wihf, WIHF, 10, 0, GATES, 300, 0);
  damic_packB<<<dim3(64, 10), 32, 0, stream>>>(wihr, WIHR, 10, 0, GATES, 300, 0);
  damic_packB<<<dim3(64, 8),  32, 0, stream>>>(whhf, WHHF, 8, 0, GATES, HID, 0);
  damic_packB<<<dim3(64, 8),  32, 0, stream>>>(whhr, WHHR, 8, 0, GATES, HID, 0);
  damic_packB<<<dim3(2, 16),  32, 0, stream>>>(headw, HWF, 16, 0, OUTD, 2 * HID, 0);

  // fused async-gather + conv + relu + maxpool
  damic_conv<<<BT, 256, 0, stream>>>(dialogue, emb, C3F, C4F, C5F,
                                     cb3, cb4, cb5, XT);
  // LSTM input projections (both directions)
  damic_xg<<<dim3(BT / 16, GATES / 16, 2), 32, 0, stream>>>(
      XT, WIHF, WIHR, bihf, bhhf, bihr, bhhr, XG);
  // recurrent scan, forward + reverse
  damic_lstm<<<2, 1024, 0, stream>>>(XG, WHHF, WHHR, HCAT);
  // classification head + sigmoid
  damic_head<<<dim3(BT / 16, OUTD / 16), 32, 0, stream>>>(
      HCAT, HWF, headb, (float*)d_out);
}
